// Propose_61409442398715
// MI455X (gfx1250) — matrix operand order repaired and checked by reading, compiled, not run on hardware
//
#include <hip/hip_runtime.h>
#include <hip/hip_bf16.h>
#include <math.h>

// ---------------------------------------------------------------------------
// Problem constants (from the reference)
// ---------------------------------------------------------------------------
#define N_NODES   8192
#define N_GRAPHS  32
#define NPG       256        // nodes per graph
#define N_EDGES   131072
#define E_TOT     (N_EDGES + N_NODES)   // self-loops appended
#define F_IN      512
#define HIDDEN    128
#define HEADS     8
#define H1_COLS   (HEADS * HIDDEN)      // 1024
#define N_CLASSES 10
#define K_POOL    205
#define NEG_SLOPE 0.2f

typedef _Float16 v16h __attribute__((ext_vector_type(16)));
typedef _Float16 v8h  __attribute__((ext_vector_type(8)));
typedef float    v8f  __attribute__((ext_vector_type(8)));

static __device__ __forceinline__ unsigned enc_f32(float f) {
    unsigned u = __float_as_uint(f);
    return (u & 0x80000000u) ? ~u : (u | 0x80000000u);
}
static __device__ __forceinline__ float dec_f32(unsigned u) {
    u = (u & 0x80000000u) ? (u & 0x7fffffffu) : ~u;
    return __uint_as_float(u);
}
// enc(-FLT_MAX): __float_as_uint(-FLT_MAX)=0xFF7FFFFF -> ~ = 0x00800000
#define ENC_NEG_FLT_MAX 0x00800000u

// ---------------------------------------------------------------------------
// Prep: W [K][N] f32 row-major  ->  Wt [N][K] f16 (K-major) for WMMA B frags
// ---------------------------------------------------------------------------
__global__ void k_transpose_cast(const float* __restrict__ W, _Float16* __restrict__ Wt,
                                 int K, int Nc) {
    int idx = blockIdx.x * blockDim.x + threadIdx.x;
    if (idx >= K * Nc) return;
    int k = idx / Nc;
    int n = idx - k * Nc;
    Wt[(size_t)n * K + k] = (_Float16)W[idx];
}

// ---------------------------------------------------------------------------
// LDS-staged, double-buffered WMMA GEMM:
//   C[M,N] = A[M,K](f32, row-major) * Bt[N,K](f16, K-major)
// Block = 256 threads = 8 waves arranged 4(m) x 2(n); block tile 64x128.
// Per K-step (Kc=32):
//   - A 64x32 tile: global f32 load -> v_cvt_pk_f16_f32 -> ds_store (needs cvt)
//   - B 128x32 tile: GLOBAL_LOAD_ASYNC_TO_LDS_B128 (ASYNCcnt path, no VGPR trip)
// LDS rows padded to 40 halves (80B) to spread banks. Double buffered, one
// barrier per K-step; s_wait_asynccnt 0 drains the async tile before barrier.
// Requires: M % 64 == 0, N % 128 == 0, K % 32 == 0.
// ---------------------------------------------------------------------------
__global__ __launch_bounds__(256)
void k_gemm_wmma_lds(const float* __restrict__ A, const _Float16* __restrict__ Bt,
                     float* __restrict__ C, int M, int N, int K) {
    __shared__ __align__(16) _Float16 As[2][64][40];
    __shared__ __align__(16) _Float16 Bs[2][128][40];

    const int nB = N >> 7;
    const int m0 = (int)(blockIdx.x / nB) << 6;
    const int n0 = (int)(blockIdx.x % nB) << 7;

    const int t    = threadIdx.x;
    const int lane = t & 31;
    const int wave = t >> 5;
    const int l16  = lane & 15;
    const int g    = lane >> 4;
    const int wm   = wave >> 1;   // 0..3  (16-row strip within 64)
    const int wn   = wave & 1;    // 0..1  (64-col strip within 128)

    // staging assignments (all 256 threads)
    const int ar = t >> 2;          // 0..63 : A row
    const int ak = (t & 3) * 8;     // 0,8,16,24 : A k-chunk (8 f32 -> 8 f16)
    const int bc = t >> 1;          // 0..127 : B col
    const int bk = (t & 1) * 16;    // 0,16 : B k-chunk (16 f16 = 32 bytes)

    const float*    gA = A  + (size_t)(m0 + ar) * K + ak;
    const _Float16* gB = Bt + (size_t)(n0 + bc) * K + bk;

    auto stage = [&](int buf, int k0) {
        // ---- A: load f32, convert, ds_store ----
        float4 f0 = *(const float4*)(gA + k0);
        float4 f1 = *(const float4*)(gA + k0 + 4);
        v8h h;
        h[0] = (_Float16)f0.x; h[1] = (_Float16)f0.y;
        h[2] = (_Float16)f0.z; h[3] = (_Float16)f0.w;
        h[4] = (_Float16)f1.x; h[5] = (_Float16)f1.y;
        h[6] = (_Float16)f1.z; h[7] = (_Float16)f1.w;
        *(v8h*)&As[buf][ar][ak] = h;
        // ---- B: async DMA global -> LDS (2 x 16B, IOFFSET applies to both) ----
        unsigned ldsB = (unsigned)(size_t)&Bs[buf][bc][bk];
        unsigned long long gaddr = (unsigned long long)(size_t)(gB + k0);
        asm volatile(
            "global_load_async_to_lds_b128 %0, %1, off\n\t"
            "global_load_async_to_lds_b128 %0, %1, off offset:16"
            :: "v"(ldsB), "v"(gaddr) : "memory");
    };

    v8f acc0 = {}, acc1 = {}, acc2 = {}, acc3 = {};

    stage(0, 0);
    asm volatile("s_wait_asynccnt 0x0" ::: "memory");
    __syncthreads();

    const int steps = K >> 5;
    for (int s = 0; s < steps; s++) {
        const int cur = s & 1;
        const int nxt = cur ^ 1;
        if (s + 2 < steps) {  // prefetch 2 K-steps ahead into GL2
            __builtin_prefetch(gA + (s + 2) * 32, 0, 1);
        }
        if (s + 1 < steps) stage(nxt, (s + 1) * 32);   // writes buffer 'nxt' only

        // ---- A fragment (contiguous k order chosen here, same as B):
        // elems 0..15 = K in [g*16, g*16+15]
        const int row = wm * 16 + l16;
        v8h alo = *(const v8h*)&As[cur][row][g * 16];
        v8h ahi = *(const v8h*)&As[cur][row][g * 16 + 8];
        v16h a = __builtin_shufflevector(alo, ahi,
                 0, 1, 2, 3, 4, 5, 6, 7, 8, 9, 10, 11, 12, 13, 14, 15);

        // ---- B fragments: lanes 0-15 hold K=0..15, lanes 16-31 hold K=16..31
        const int colb = wn * 64 + l16;
        v8h b0lo = *(const v8h*)&Bs[cur][colb +  0][g * 16];
        v8h b0hi = *(const v8h*)&Bs[cur][colb +  0][g * 16 + 8];
        v8h b1lo = *(const v8h*)&Bs[cur][colb + 16][g * 16];
        v8h b1hi = *(const v8h*)&Bs[cur][colb + 16][g * 16 + 8];
        v8h b2lo = *(const v8h*)&Bs[cur][colb + 32][g * 16];
        v8h b2hi = *(const v8h*)&Bs[cur][colb + 32][g * 16 + 8];
        v8h b3lo = *(const v8h*)&Bs[cur][colb + 48][g * 16];
        v8h b3hi = *(const v8h*)&Bs[cur][colb + 48][g * 16 + 8];
        v16h b0 = __builtin_shufflevector(b0lo, b0hi,
                  0, 1, 2, 3, 4, 5, 6, 7, 8, 9, 10, 11, 12, 13, 14, 15);
        v16h b1 = __builtin_shufflevector(b1lo, b1hi,
                  0, 1, 2, 3, 4, 5, 6, 7, 8, 9, 10, 11, 12, 13, 14, 15);
        v16h b2 = __builtin_shufflevector(b2lo, b2hi,
                  0, 1, 2, 3, 4, 5, 6, 7, 8, 9, 10, 11, 12, 13, 14, 15);
        v16h b3 = __builtin_shufflevector(b3lo, b3hi,
                  0, 1, 2, 3, 4, 5, 6, 7, 8, 9, 10, 11, 12, 13, 14, 15);

        acc0 = __builtin_amdgcn_wmma_f32_16x16x32_f16(false, a, false, b0, (short)0, acc0, false, false);
        acc1 = __builtin_amdgcn_wmma_f32_16x16x32_f16(false, a, false, b1, (short)0, acc1, false, false);
        acc2 = __builtin_amdgcn_wmma_f32_16x16x32_f16(false, a, false, b2, (short)0, acc2, false, false);
        acc3 = __builtin_amdgcn_wmma_f32_16x16x32_f16(false, a, false, b3, (short)0, acc3, false, false);

        // drain this iteration's async B-tile writes, then block-wide barrier
        asm volatile("s_wait_asynccnt 0x0" ::: "memory");
        __syncthreads();
    }

    // ---- store (32-bit C/D layout: VGPR v -> row v + 8*g, col = lane&15)
    float* cp = C + (size_t)(m0 + wm * 16 + g * 8) * N + n0 + wn * 64 + l16;
    #pragma unroll
    for (int v = 0; v < 8; v++) {
        cp[(size_t)v * N + 0]  = acc0[v];
        cp[(size_t)v * N + 16] = acc1[v];
        cp[(size_t)v * N + 32] = acc2[v];
        cp[(size_t)v * N + 48] = acc3[v];
    }
}

// ---------------------------------------------------------------------------
// Attention scores: wave per (node, head); C = 128 fixed
// ---------------------------------------------------------------------------
__global__ __launch_bounds__(256)
void k_attn_scores(const float* __restrict__ Hb, const float* __restrict__ att_s,
                   const float* __restrict__ att_d, float* __restrict__ asrc,
                   float* __restrict__ adst, int Hn, int hshift) {
    int gw   = (int)((blockIdx.x * blockDim.x + threadIdx.x) >> 5);
    int lane = threadIdx.x & 31;
    int n = gw >> hshift;
    int h = gw & (Hn - 1);
    const float* hp = Hb + ((size_t)n * Hn + h) * HIDDEN + lane * 4;
    const float* sp = att_s + h * HIDDEN + lane * 4;
    const float* dp = att_d + h * HIDDEN + lane * 4;
    float ss = 0.f, dd = 0.f;
    #pragma unroll
    for (int i = 0; i < 4; i++) { float hv = hp[i]; ss += hv * sp[i]; dd += hv * dp[i]; }
    #pragma unroll
    for (int m = 16; m >= 1; m >>= 1) {
        ss += __shfl_xor(ss, m, 32);
        dd += __shfl_xor(dd, m, 32);
    }
    if (lane == 0) {
        asrc[(size_t)n * Hn + h] = ss;
        adst[(size_t)n * Hn + h] = dd;
    }
}

// ---------------------------------------------------------------------------
// Edge pass 1: e = leaky_relu(asrc[src]+adst[dst]); segment max via atomicMax
// ---------------------------------------------------------------------------
__global__ __launch_bounds__(256)
void k_edge_logit_max(const int* __restrict__ src, const int* __restrict__ dst,
                      const float* __restrict__ asrc, const float* __restrict__ adst,
                      float* __restrict__ ebuf, unsigned* __restrict__ mbuf,
                      int E0, long long total, int Hn, int hshift) {
    long long idx = (long long)blockIdx.x * blockDim.x + threadIdx.x;
    if (idx >= total) return;
    int e = (int)(idx >> hshift);
    int h = (int)(idx & (Hn - 1));
    int s_ = (e < E0) ? src[e] : (e - E0);
    int d_ = (e < E0) ? dst[e] : (e - E0);
    float v = asrc[(size_t)s_ * Hn + h] + adst[(size_t)d_ * Hn + h];
    v = (v > 0.f) ? v : NEG_SLOPE * v;
    ebuf[idx] = v;
    atomicMax(&mbuf[(size_t)d_ * Hn + h], enc_f32(v));
}

// ---------------------------------------------------------------------------
// Edge pass 2: ex = exp(e - m[dst]); segment sum via atomicAdd; ebuf <- ex
// ---------------------------------------------------------------------------
__global__ __launch_bounds__(256)
void k_edge_exp_sum(const int* __restrict__ src, const int* __restrict__ dst,
                    float* __restrict__ ebuf, const unsigned* __restrict__ mbuf,
                    float* __restrict__ sbuf, int E0, long long total,
                    int Hn, int hshift) {
    long long idx = (long long)blockIdx.x * blockDim.x + threadIdx.x;
    if (idx >= total) return;
    int e = (int)(idx >> hshift);
    int h = (int)(idx & (Hn - 1));
    int d_ = (e < E0) ? dst[e] : (e - E0);
    float m = dec_f32(mbuf[(size_t)d_ * Hn + h]);
    float ex = expf(ebuf[idx] - m);
    ebuf[idx] = ex;
    atomicAdd(&sbuf[(size_t)d_ * Hn + h], ex);
}

// ---------------------------------------------------------------------------
// Edge pass 3: O[dst,h,:] += (ex/s[dst,h]) * H[src,h,:]
// thread per (edge, head, channel); warp = 32 consecutive channels
// ---------------------------------------------------------------------------
__global__ __launch_bounds__(256)
void k_edge_aggregate(const int* __restrict__ src, const int* __restrict__ dst,
                      const float* __restrict__ ebuf, const float* __restrict__ sbuf,
                      const float* __restrict__ Hb, float* __restrict__ O,
                      int E0, long long total, int Hn, int hshift) {
    long long idx = (long long)blockIdx.x * blockDim.x + threadIdx.x;
    if (idx >= total) return;
    int c  = (int)(idx & (HIDDEN - 1));
    long long eh = idx >> 7;            // (e*Hn + h)
    int e = (int)(eh >> hshift);
    int h = (int)(eh & (Hn - 1));
    int s_ = (e < E0) ? src[e] : (e - E0);
    int d_ = (e < E0) ? dst[e] : (e - E0);
    float alpha = ebuf[eh] / (sbuf[(size_t)d_ * Hn + h] + 1e-16f);
    float contrib = alpha * Hb[((size_t)s_ * Hn + h) * HIDDEN + c];
    atomicAdd(&O[((size_t)d_ * Hn + h) * HIDDEN + c], contrib);
}

// ---------------------------------------------------------------------------
// Utility kernels
// ---------------------------------------------------------------------------
__global__ void k_fill_u32(unsigned* __restrict__ p, unsigned v, long long n) {
    long long i = (long long)blockIdx.x * blockDim.x + threadIdx.x;
    if (i < n) p[i] = v;
}

__global__ void k_bias_relu(float* __restrict__ O, const float* __restrict__ b,
                            long long n, int cmask) {
    long long i = (long long)blockIdx.x * blockDim.x + threadIdx.x;
    if (i >= n) return;
    float v = O[i] + b[i & cmask];
    O[i] = (v > 0.f) ? v : 0.f;
}

// score[n] = tanh(dot(O2[n], w) / ||w||); wave per node
__global__ __launch_bounds__(256)
void k_score(const float* __restrict__ O2, const float* __restrict__ pw,
             float* __restrict__ score) {
    int n    = (int)((blockIdx.x * blockDim.x + threadIdx.x) >> 5);
    int lane = threadIdx.x & 31;
    const float* op = O2 + (size_t)n * HIDDEN + lane * 4;
    const float* wp = pw + lane * 4;
    float dot = 0.f, wn = 0.f;
    #pragma unroll
    for (int i = 0; i < 4; i++) { float w = wp[i]; dot += op[i] * w; wn += w * w; }
    #pragma unroll
    for (int m = 16; m >= 1; m >>= 1) {
        dot += __shfl_xor(dot, m, 32);
        wn  += __shfl_xor(wn,  m, 32);
    }
    if (lane == 0) score[n] = tanhf(dot * rsqrtf(wn));
}

// TopK (k=205 of 256) per graph via stable rank; weighted mean -> pooled[g][c]
__global__ __launch_bounds__(256)
void k_topk_pool(const float* __restrict__ O2, const float* __restrict__ score,
                 float* __restrict__ pooled) {
    __shared__ float sc[NPG];
    __shared__ float coef[NPG];
    int g = blockIdx.x;
    int t = threadIdx.x;
    float s = score[g * NPG + t];
    sc[t] = s;
    __syncthreads();
    int rank = 0;
    for (int j = 0; j < NPG; j++) {
        float o = sc[j];
        rank += (o > s) || (o == s && j < t);   // stable tie-break by index
    }
    coef[t] = (rank < K_POOL) ? s : 0.f;
    __syncthreads();
    if (t < HIDDEN) {
        float acc = 0.f;
        for (int n = 0; n < NPG; n++)
            acc += coef[n] * O2[((size_t)(g * NPG + n)) * HIDDEN + t];
        pooled[g * HIDDEN + t] = acc * (1.0f / (float)K_POOL);
    }
}

// logits = pooled @ lin_w + lin_b; log_softmax; one wave per graph
__global__ __launch_bounds__(32)
void k_head(const float* __restrict__ pooled, const float* __restrict__ lin_w,
            const float* __restrict__ lin_b, float* __restrict__ out) {
    __shared__ float lg[N_CLASSES];
    int b = blockIdx.x;
    int t = threadIdx.x;
    if (t < N_CLASSES) {
        float acc = lin_b[t];
        for (int c = 0; c < HIDDEN; c++)
            acc += pooled[b * HIDDEN + c] * lin_w[c * N_CLASSES + t];
        lg[t] = acc;
    }
    __syncthreads();
    if (t < N_CLASSES) {
        float mx = lg[0];
        #pragma unroll
        for (int i = 1; i < N_CLASSES; i++) mx = fmaxf(mx, lg[i]);
        float se = 0.f;
        #pragma unroll
        for (int i = 0; i < N_CLASSES; i++) se += expf(lg[i] - mx);
        out[b * N_CLASSES + t] = lg[t] - mx - logf(se);
    }
}

// ---------------------------------------------------------------------------
// Host orchestration
// ---------------------------------------------------------------------------
static inline long long cdiv_ll(long long a, long long b) { return (a + b - 1) / b; }

extern "C" void kernel_launch(void* const* d_in, const int* in_sizes, int n_in,
                              void* d_out, int out_size, void* d_ws, size_t ws_size,
                              hipStream_t stream) {
    (void)in_sizes; (void)n_in; (void)out_size; (void)ws_size;

    const float* x        = (const float*)d_in[0];
    const int*   ei       = (const int*)d_in[1];
    const int*   src      = ei;
    const int*   dst      = ei + N_EDGES;
    const float* W1       = (const float*)d_in[3];
    const float* att_src1 = (const float*)d_in[4];
    const float* att_dst1 = (const float*)d_in[5];
    const float* b1       = (const float*)d_in[6];
    const float* W2       = (const float*)d_in[7];
    const float* att_src2 = (const float*)d_in[8];
    const float* att_dst2 = (const float*)d_in[9];
    const float* b2       = (const float*)d_in[10];
    const float* pool_w   = (const float*)d_in[11];
    const float* lin_w    = (const float*)d_in[12];
    const float* lin_b    = (const float*)d_in[13];
    float* out = (float*)d_out;

    // ---- carve workspace ----
    char* ws = (char*)d_ws;
    size_t off = 0;
    auto carve = [&](size_t bytes) -> char* {
        char* p = ws + off;
        off += (bytes + 255) & ~(size_t)255;
        return p;
    };
    float*    H1    = (float*)   carve((size_t)N_NODES * H1_COLS * 4);   // 32 MB
    float*    O1    = (float*)   carve((size_t)N_NODES * H1_COLS * 4);   // 32 MB
    _Float16* W1t   = (_Float16*)carve((size_t)F_IN * H1_COLS * 2);      // 1 MB
    _Float16* W2t   = (_Float16*)carve((size_t)H1_COLS * HIDDEN * 2);    // 0.25 MB
    float*    asrc1 = (float*)   carve((size_t)N_NODES * HEADS * 4);
    float*    adst1 = (float*)   carve((size_t)N_NODES * HEADS * 4);
    unsigned* m1    = (unsigned*)carve((size_t)N_NODES * HEADS * 4);
    float*    s1    = (float*)   carve((size_t)N_NODES * HEADS * 4);
    float*    eb1   = (float*)   carve((size_t)E_TOT * HEADS * 4);       // 4.5 MB
    float*    H2    = (float*)   carve((size_t)N_NODES * HIDDEN * 4);    // 4 MB
    float*    O2    = (float*)   carve((size_t)N_NODES * HIDDEN * 4);    // 4 MB
    float*    asrc2 = (float*)   carve((size_t)N_NODES * 4);
    float*    adst2 = (float*)   carve((size_t)N_NODES * 4);
    unsigned* m2    = (unsigned*)carve((size_t)N_NODES * 4);
    float*    s2    = (float*)   carve((size_t)N_NODES * 4);
    float*    eb2   = (float*)   carve((size_t)E_TOT * 4);
    float*    score = (float*)   carve((size_t)N_NODES * 4);
    float*    pooled= (float*)   carve((size_t)N_GRAPHS * HIDDEN * 4);

    const int B = 256;

    // ---- prep weights ----
    {
        long long n1 = (long long)F_IN * H1_COLS;
        k_transpose_cast<<<(unsigned)cdiv_ll(n1, B), B, 0, stream>>>(W1, W1t, F_IN, H1_COLS);
        long long n2 = (long long)H1_COLS * HIDDEN;
        k_transpose_cast<<<(unsigned)cdiv_ll(n2, B), B, 0, stream>>>(W2, W2t, H1_COLS, HIDDEN);
    }

    // ---- GEMM1: H1 = x @ W1  (8192 x 512 x 1024), block tile 64x128 ----
    {
        unsigned blocks = (N_NODES / 64) * (H1_COLS / 128);   // 128 * 8 = 1024
        k_gemm_wmma_lds<<<blocks, B, 0, stream>>>(x, W1t, H1, N_NODES, H1_COLS, F_IN);
    }

    // ---- layer 1 attention ----
    {
        long long nh = (long long)N_NODES * HEADS;
        k_fill_u32<<<(unsigned)cdiv_ll(nh, B), B, 0, stream>>>(m1, ENC_NEG_FLT_MAX, nh);
        k_fill_u32<<<(unsigned)cdiv_ll(nh, B), B, 0, stream>>>((unsigned*)s1, 0u, nh);
        long long no = (long long)N_NODES * H1_COLS;
        k_fill_u32<<<(unsigned)cdiv_ll(no, B), B, 0, stream>>>((unsigned*)O1, 0u, no);

        k_attn_scores<<<(unsigned)(nh * 32 / B), B, 0, stream>>>(H1, att_src1, att_dst1,
                                                                 asrc1, adst1, HEADS, 3);
        long long te = (long long)E_TOT * HEADS;
        k_edge_logit_max<<<(unsigned)cdiv_ll(te, B), B, 0, stream>>>(src, dst, asrc1, adst1,
                                                                     eb1, m1, N_EDGES, te, HEADS, 3);
        k_edge_exp_sum<<<(unsigned)cdiv_ll(te, B), B, 0, stream>>>(src, dst, eb1, m1, s1,
                                                                   N_EDGES, te, HEADS, 3);
        long long ta = te * HIDDEN;
        k_edge_aggregate<<<(unsigned)cdiv_ll(ta, B), B, 0, stream>>>(src, dst, eb1, s1, H1, O1,
                                                                     N_EDGES, ta, HEADS, 3);
        k_bias_relu<<<(unsigned)cdiv_ll(no, B), B, 0, stream>>>(O1, b1, no, H1_COLS - 1);
    }

    // ---- GEMM2: H2 = O1 @ W2  (8192 x 1024 x 128), block tile 64x128 ----
    {
        unsigned blocks = (N_NODES / 64) * (HIDDEN / 128);    // 128
        k_gemm_wmma_lds<<<blocks, B, 0, stream>>>(O1, W2t, H2, N_NODES, HIDDEN, H1_COLS);
    }

    // ---- layer 2 attention (1 head) ----
    {
        long long nh = (long long)N_NODES;
        k_fill_u32<<<(unsigned)cdiv_ll(nh, B), B, 0, stream>>>(m2, ENC_NEG_FLT_MAX, nh);
        k_fill_u32<<<(unsigned)cdiv_ll(nh, B), B, 0, stream>>>((unsigned*)s2, 0u, nh);
        long long no = (long long)N_NODES * HIDDEN;
        k_fill_u32<<<(unsigned)cdiv_ll(no, B), B, 0, stream>>>((unsigned*)O2, 0u, no);

        k_attn_scores<<<(unsigned)(nh * 32 / B), B, 0, stream>>>(H2, att_src2, att_dst2,
                                                                 asrc2, adst2, 1, 0);
        long long te = (long long)E_TOT;
        k_edge_logit_max<<<(unsigned)cdiv_ll(te, B), B, 0, stream>>>(src, dst, asrc2, adst2,
                                                                     eb2, m2, N_EDGES, te, 1, 0);
        k_edge_exp_sum<<<(unsigned)cdiv_ll(te, B), B, 0, stream>>>(src, dst, eb2, m2, s2,
                                                                   N_EDGES, te, 1, 0);
        long long ta = te * HIDDEN;
        k_edge_aggregate<<<(unsigned)cdiv_ll(ta, B), B, 0, stream>>>(src, dst, eb2, s2, H2, O2,
                                                                     N_EDGES, ta, 1, 0);
        k_bias_relu<<<(unsigned)cdiv_ll(no, B), B, 0, stream>>>(O2, b2, no, HIDDEN - 1);
    }

    // ---- TopK pooling + head ----
    k_score<<<(unsigned)((long long)N_NODES * 32 / B), B, 0, stream>>>(O2, pool_w, score);
    k_topk_pool<<<N_GRAPHS, NPG, 0, stream>>>(O2, score, pooled);
    k_head<<<N_GRAPHS, 32, 0, stream>>>(pooled, lin_w, lin_b, out);
}
